// AdaptiveGraphNetwork_58188216926643
// MI455X (gfx1250) — compile-verified, matrix operand order checked by reading, not run on hardware
//
#include <hip/hip_runtime.h>
#include <hip/hip_bf16.h>
#include <stdint.h>

// CDNA5 / gfx1250, wave32. Chain-GNN restructured into dense f16 WMMA GEMMs.
// One workgroup = 4 waves = 128 threads, processes BPW batches sequentially.
//   GEMM1: x[64x32] @ [Wm1_top|Wm1_bot][32x128] -> AB[64x128]      (32 wmma)
//   h1 top/bot [64x64] built elementwise from AB (+bm1, relu, edge masks)
//   GEMM2: h1[64x64] @ Wm2[64x32] twice (top/bot)                   (32 wmma)
//     combine: messages = maskrelu(top+bm2) + maskrelu(bot+bm2)
//   GEMM3: [x|messages][64x64] @ Wu1[64x64] -> relu(+bu1) -> Ch     (32 wmma)
//   GEMM4: Ch[64x64] @ Wu2[64x32] (+bu2), residual with fp32 x      (16 wmma)
// Wm2/Wu2 WMMA B-fragments are loop-invariant -> built once in registers.
// fp32 x staged in LDS so the residual epilogue does no extra HBM reads.

typedef __attribute__((ext_vector_type(16))) _Float16 v16h;
typedef __attribute__((ext_vector_type(8)))  _Float16 v8h;
typedef __attribute__((ext_vector_type(4)))  _Float16 v4h;
typedef __attribute__((ext_vector_type(8)))  float    v8f;

#define BPW 4          // batches per workgroup
#define NB  4096       // total batches
#define NJ  64         // joints
#define DD  32         // feature dim
#define HH  64         // hidden dim

// Load a 16x32 f16 fragment (A or B^T) from row-major LDS.
// ISA layout (16-bit A 16x32): lane l holds row (l&15); k-groups
// {k0..k0+7, k0+16..k0+23}, k0 = (l>=16) ? 8 : 0.  Two ds_load_b128 per lane.
__device__ __forceinline__ v16h load_frag(const _Float16* base, int stride) {
    const int l  = threadIdx.x & 31;
    const int r  = l & 15;
    const int k0 = (l & 16) ? 8 : 0;
    const _Float16* p = base + r * stride + k0;
    v8h lo = *(const v8h*)(p);
    v8h hi = *(const v8h*)(p + 16);
    return __builtin_shufflevector(lo, hi, 0,1,2,3,4,5,6,7,8,9,10,11,12,13,14,15);
}

// Build a B-fragment (column 'col' group) directly from a row-major fp32
// global weight matrix W[.][ncols]; k range = kbase..kbase+31. One-time cost.
__device__ __forceinline__ v16h load_bfrag_global(const float* __restrict__ W,
                                                  int ncols, int col, int kbase) {
    const int l  = threadIdx.x & 31;
    const int k0 = (l & 16) ? 8 : 0;
    v16h r;
#pragma unroll
    for (int i = 0; i < 8; ++i) {
        r[i]     = (_Float16)W[(kbase + k0 + i) * ncols + col];
        r[i + 8] = (_Float16)W[(kbase + k0 + 16 + i) * ncols + col];
    }
    return r;
}

__device__ __forceinline__ v8f wmma_f16(v16h a, v16h b, v8f c) {
    return __builtin_amdgcn_wmma_f32_16x16x32_f16(
        /*neg_a=*/false, a, /*neg_b=*/false, b,
        /*c_mod=*/(short)0, c, /*reuse_a=*/false, /*reuse_b=*/false);
}

// Store a 16x16 f32 D-tile as f16 into row-major LDS.
// D layout: lane l -> col n=(l&15), rows m = 8*(l>>4)+v.
__device__ __forceinline__ void store_d_f16(v8f d, _Float16* base, int stride) {
    const int l  = threadIdx.x & 31;
    const int n  = l & 15;
    const int mb = (l >> 4) * 8;
#pragma unroll
    for (int v = 0; v < 8; ++v)
        base[(mb + v) * stride + n] = (_Float16)d[v];
}

__global__ __launch_bounds__(128) void agn_kernel(
    const float* __restrict__ x,      // [NB, NJ, DD]
    const float* __restrict__ Wm1,    // [2*DD, HH]
    const float* __restrict__ bm1,    // [HH]
    const float* __restrict__ Wm2,    // [HH, DD]
    const float* __restrict__ bm2,    // [DD]
    const float* __restrict__ Wu1,    // [2*DD, HH]
    const float* __restrict__ bu1,    // [HH]
    const float* __restrict__ Wu2,    // [HH, DD]
    const float* __restrict__ bu2,    // [DD]
    const float* __restrict__ resw,   // scalar
    float* __restrict__ out)          // [NB, NJ, DD]
{
    // ---- shared memory (~57KB total) ----
    __shared__ __align__(16) _Float16 sWm1T[64 * 64];  // [h][k] = Wm1[k][h]
    __shared__ __align__(16) _Float16 sWu1T[64 * 64];  // [h][k] = Wu1[k][h]
    __shared__ float sbm1[64], sbm2[32], sbu1[64], sbu2[32];
    __shared__ __align__(16) _Float16 sUpd[64 * 64];   // cols 0..31: x(f16), 32..63: messages
    __shared__ __align__(16) _Float16 sAB [64 * 128];  // GEMM1 out; later reused as Ch[64][64]
    __shared__ __align__(16) _Float16 sH1 [64 * 64];   // edge hidden, top then bottom
    __shared__ __align__(16) float    sXf [64 * 32];   // fp32 x stage (residual source)

    const int tid = threadIdx.x;
    const int w   = tid >> 5;           // wave id 0..3
    const int l   = tid & 31;
    const int n16 = l & 15;
    const int mb  = (l >> 4) * 8;

    // ---- loop-invariant register B-fragments for Wm2 / Wu2 (64 VGPRs) ----
    v16h rB2 [2][2], rBu2[2][2];        // [j][kb]
#pragma unroll
    for (int j = 0; j < 2; ++j)
#pragma unroll
        for (int kb = 0; kb < 2; ++kb) {
            rB2 [j][kb] = load_bfrag_global(Wm2, DD, 16 * j + n16, kb * 32);
            rBu2[j][kb] = load_bfrag_global(Wu2, DD, 16 * j + n16, kb * 32);
        }

    // ---- stage weights (transposed, f16) + biases into LDS ----
    for (int e = tid; e < 64 * 64; e += 128) {
        int k = e >> 6, h = e & 63;
        sWm1T[h * 64 + k] = (_Float16)Wm1[e];
        sWu1T[h * 64 + k] = (_Float16)Wu1[e];
    }
    if (tid < 64) { sbm1[tid] = bm1[tid]; sbu1[tid] = bu1[tid]; }
    if (tid < 32) { sbm2[tid] = bm2[tid]; sbu2[tid] = bu2[tid]; }
    const float rw = resw[0];
    __syncthreads();

    for (int it = 0; it < BPW; ++it) {
        const int b = blockIdx.x * BPW + it;
        const float* xb = x + (size_t)b * (NJ * DD);

        // speculative prefetch of next batch's features
        if (it + 1 < BPW)
            __builtin_prefetch(xb + NJ * DD + tid * 16, 0, 1);

        // ---- load x: fp32 -> sXf stage, f16 -> sUpd cols 0..31 ----
        for (int e4 = tid; e4 < (NJ * DD) / 4; e4 += 128) {
            float4 v = ((const float4*)xb)[e4];
            ((float4*)sXf)[e4] = v;
            int e = e4 * 4;
            int r = e >> 5, c = e & 31;
            v4h h; h[0] = (_Float16)v.x; h[1] = (_Float16)v.y;
                   h[2] = (_Float16)v.z; h[3] = (_Float16)v.w;
            *(v4h*)(sUpd + r * 64 + c) = h;
        }
        __syncthreads();

        // ---- GEMM1: wave w does row-tile i=w, all 8 col-tiles ----
        {
            v16h a = load_frag(sUpd + (16 * w) * 64, 64);  // rows 16w.., K=0..31
#pragma unroll
            for (int j = 0; j < 8; ++j) {
                const _Float16* bt = (j < 4)
                    ? (sWm1T + (16 * j) * 64)              // top half of Wm1
                    : (sWm1T + (16 * (j - 4)) * 64 + 32);  // bottom half
                v16h bf = load_frag(bt, 64);
                v8f c = {};
                c = wmma_f16(a, bf, c);
                store_d_f16(c, sAB + (16 * w) * 128 + 16 * j, 128);
            }
        }
        __syncthreads();

        // ---- h1 top: row n = edge(dst=n, src=n+1), valid n<63 ----
        for (int e = tid; e < 64 * 64; e += 128) {
            int n = e >> 6, h = e & 63;
            float v = 0.f;
            if (n < 63) {
                v = (float)sAB[n * 128 + h] + (float)sAB[(n + 1) * 128 + 64 + h] + sbm1[h];
                v = v > 0.f ? v : 0.f;
            }
            sH1[e] = (_Float16)v;
        }
        __syncthreads();

        // ---- GEMM2a (top messages, kept in VGPRs) ----
        v8f p0 = {}, p1 = {};
#pragma unroll
        for (int kb = 0; kb < 2; ++kb) {
            v16h a = load_frag(sH1 + (16 * w) * 64 + kb * 32, 64);
            p0 = wmma_f16(a, rB2[0][kb], p0);
            p1 = wmma_f16(a, rB2[1][kb], p1);
        }
        __syncthreads();  // everyone done reading h1-top before overwrite

        // ---- h1 bottom: row n = edge(dst=n, src=n-1), valid n>0 ----
        for (int e = tid; e < 64 * 64; e += 128) {
            int n = e >> 6, h = e & 63;
            float v = 0.f;
            if (n > 0) {
                v = (float)sAB[n * 128 + h] + (float)sAB[(n - 1) * 128 + 64 + h] + sbm1[h];
                v = v > 0.f ? v : 0.f;
            }
            sH1[e] = (_Float16)v;
        }
        __syncthreads();

        // ---- GEMM2b + combine -> messages into sUpd cols 32..63 ----
        v8f q0 = {}, q1 = {};
#pragma unroll
        for (int kb = 0; kb < 2; ++kb) {
            v16h a = load_frag(sH1 + (16 * w) * 64 + kb * 32, 64);
            q0 = wmma_f16(a, rB2[0][kb], q0);
            q1 = wmma_f16(a, rB2[1][kb], q1);
        }
#pragma unroll
        for (int v = 0; v < 8; ++v) {
            int m = 16 * w + mb + v;
            float t0 = p0[v] + sbm2[n16];
            float t1 = p1[v] + sbm2[16 + n16];
            float u0 = q0[v] + sbm2[n16];
            float u1 = q1[v] + sbm2[16 + n16];
            float top0 = (m < 63) ? (t0 > 0.f ? t0 : 0.f) : 0.f;   // edge mask: dst=63 has no (src=64)
            float top1 = (m < 63) ? (t1 > 0.f ? t1 : 0.f) : 0.f;
            float bot0 = (m > 0)  ? (u0 > 0.f ? u0 : 0.f) : 0.f;   // edge mask: dst=0 has no (src=-1)
            float bot1 = (m > 0)  ? (u1 > 0.f ? u1 : 0.f) : 0.f;
            sUpd[m * 64 + 32 + n16]      = (_Float16)(top0 + bot0);
            sUpd[m * 64 + 32 + 16 + n16] = (_Float16)(top1 + bot1);
        }
        __syncthreads();

        // ---- GEMM3: [x|messages] @ Wu1, relu(+bu1) -> Ch (reuses sAB) ----
        _Float16* Ch = sAB;  // [64][64], sAB is dead now
        {
            v16h a0 = load_frag(sUpd + (16 * w) * 64 + 0,  64);
            v16h a1 = load_frag(sUpd + (16 * w) * 64 + 32, 64);
#pragma unroll
            for (int j = 0; j < 4; ++j) {
                v8f c = {};
                c = wmma_f16(a0, load_frag(sWu1T + (16 * j) * 64 + 0,  64), c);
                c = wmma_f16(a1, load_frag(sWu1T + (16 * j) * 64 + 32, 64), c);
#pragma unroll
                for (int v = 0; v < 8; ++v) {
                    float t = c[v] + sbu1[16 * j + n16];
                    Ch[(16 * w + mb + v) * 64 + 16 * j + n16] = (_Float16)(t > 0.f ? t : 0.f);
                }
            }
        }
        // same-wave DS ops are in-order: GEMM4 may read Ch rows 16w immediately

        // ---- GEMM4 + residual epilogue (x residual from LDS fp32 stage) ----
        {
            v16h a0 = load_frag(Ch + (16 * w) * 64 + 0,  64);
            v16h a1 = load_frag(Ch + (16 * w) * 64 + 32, 64);
#pragma unroll
            for (int j = 0; j < 2; ++j) {
                v8f c = {};
                c = wmma_f16(a0, rBu2[j][0], c);
                c = wmma_f16(a1, rBu2[j][1], c);
#pragma unroll
                for (int v = 0; v < 8; ++v) {
                    int m   = 16 * w + mb + v;
                    int col = 16 * j + n16;
                    size_t gi = (size_t)b * (NJ * DD) + (size_t)m * DD + col;
                    float upd = c[v] + sbu2[col];
                    out[gi] = rw * upd + (1.f - rw) * sXf[m * DD + col];
                }
            }
        }
        __syncthreads();  // protect sUpd/sAB/sXf reuse next iteration
    }
}

extern "C" void kernel_launch(void* const* d_in, const int* in_sizes, int n_in,
                              void* d_out, int out_size, void* d_ws, size_t ws_size,
                              hipStream_t stream) {
    (void)in_sizes; (void)n_in; (void)out_size; (void)d_ws; (void)ws_size;
    const float* x    = (const float*)d_in[0];
    const float* Wm1  = (const float*)d_in[1];
    const float* bm1  = (const float*)d_in[2];
    const float* Wm2  = (const float*)d_in[3];
    const float* bm2  = (const float*)d_in[4];
    const float* Wu1  = (const float*)d_in[5];
    const float* bu1  = (const float*)d_in[6];
    const float* Wu2  = (const float*)d_in[7];
    const float* bu2  = (const float*)d_in[8];
    const float* resw = (const float*)d_in[9];
    float* out = (float*)d_out;

    dim3 grid(NB / BPW);   // 1024 workgroups, 4 batches each
    dim3 block(128);       // 4 waves (wave32)
    agn_kernel<<<grid, block, 0, stream>>>(x, Wm1, bm1, Wm2, bm2,
                                           Wu1, bu1, Wu2, bu2, resw, out);
}
